// thetaRNNLayer_420906795706
// MI455X (gfx1250) — matrix operand order repaired and checked by reading, not compiled, not run on hardware
//
#include <hip/hip_runtime.h>

// ---------------------------------------------------------------------------
// thetaRNN layer for MI455X (gfx1250, wave32, WMMA bf16 w/ f32 accumulate)
//   pre = input @ W_ih^T + internal + bias            (one big WMMA GEMM)
//   for t: h = tanh(pre[:,t,:] + h @ W_hh^T)          (512 step kernels)
// A-strips are staged through LDS once per block (8 waves share the strip).
// ---------------------------------------------------------------------------

typedef __attribute__((ext_vector_type(16))) __bf16 v16bf;
typedef __attribute__((ext_vector_type(8)))  float  v8f;

union FragBF {
    v16bf        v;
    unsigned int u[8];
};

// LDS row stride in dwords: 512 bf16 = 256 dwords, +4 pad dwords so the 16
// lanes of a half-wave land on banks 0,4,8,...,60 (260 % 64 == 4) and each
// ds_load_b128 covers 4 consecutive banks -> conflict-free.
#define LDS_STRIDE 260

__device__ __forceinline__ unsigned int pack2_bf16(float lo, float hi) {
    // round-to-nearest-even fp32 -> bf16, packed pair (low half = first elem)
    unsigned int a = __float_as_uint(lo);
    unsigned int b = __float_as_uint(hi);
    a += 0x7FFFu + ((a >> 16) & 1u);
    b += 0x7FFFu + ((b >> 16) & 1u);
    return (a >> 16) | (b & 0xFFFF0000u);
}

__device__ __forceinline__ unsigned short f32_to_bf16(float f) {
    unsigned int a = __float_as_uint(f);
    a += 0x7FFFu + ((a >> 16) & 1u);
    return (unsigned short)(a >> 16);
}

// ---------------------------------------------------------------------------
// elementwise fp32 -> bf16 conversion (weights, initial hidden state)
// ---------------------------------------------------------------------------
__global__ void __launch_bounds__(256)
cvt_f32_bf16(const float* __restrict__ src, unsigned short* __restrict__ dst, int n) {
    int i = blockIdx.x * 256 + threadIdx.x;
    if (i < n) dst[i] = f32_to_bf16(src[i]);
}

// ---------------------------------------------------------------------------
// Phase 1: pre[m, n] = sum_k input[m,k] * W_ih[n,k] + internal[m,n] + bias[n]
//   M = B*S = 32768, N = H = 512, K = I = 512
//   1 block = 8 waves; block owns a 16-row M strip, wave covers 64 N columns.
//   A strip (16x512 fp32) is loaded+converted to bf16 in LDS once per block,
//   then each wave pulls WMMA A fragments with ds_load_b128.
//   grid = M/16 = 2048 blocks.
// ---------------------------------------------------------------------------
__global__ void __launch_bounds__(256)
rnn_pre_gemm(const float* __restrict__ X,                 // [M, K] fp32
             const unsigned short* __restrict__ Wih,      // [N, K] bf16 bits
             const float* __restrict__ internal,          // [M, N]
             const float* __restrict__ bias,              // [N]
             float* __restrict__ pre)                     // [M, N] (d_out)
{
    constexpr int K = 512, N = 512;
    __shared__ unsigned int lds_a[16 * LDS_STRIDE];       // 16 rows bf16, padded

    const int tid    = threadIdx.x;
    const int wave   = tid >> 5;
    const int lane   = tid & 31;
    const int half   = lane >> 4;        // 0: lanes 0-15, 1: lanes 16-31
    const int lanelo = lane & 15;
    const int m0     = blockIdx.x * 16;
    const int n0     = wave * 64;

    // ---- cooperative stage: 16x512 fp32 -> bf16 LDS (2048 float4 total) ----
    for (int q = tid; q < 2048; q += 256) {
        const int row = q >> 7;              // 128 float4 per row
        const int c4  = (q & 127) * 4;       // starting column
        const float4 x = *(const float4*)(X + (size_t)(m0 + row) * K + c4);
        const int d = row * LDS_STRIDE + (c4 >> 1);
        lds_a[d]     = pack2_bf16(x.x, x.y);
        lds_a[d + 1] = pack2_bf16(x.z, x.w);
    }
    __syncthreads();

    v8f zero = {0.f, 0.f, 0.f, 0.f, 0.f, 0.f, 0.f, 0.f};
    v8f acc[4] = {zero, zero, zero, zero};

    const unsigned int* alds = lds_a + lanelo * LDS_STRIDE;

    for (int kb = 0; kb < K; kb += 32) {
        // A fragment: lane half 0 -> K run {kb..kb+15}
        //             lane half 1 -> K run {kb+16..kb+31}
        const int di = (kb >> 1) + half * 8;          // dword offset of run0
        FragBF a;
        uint4 alo = *(const uint4*)(alds + di);       // run0: 8 bf16
        uint4 ahi = *(const uint4*)(alds + di + 4);   // run1: +8 bf16
        a.u[0] = alo.x; a.u[1] = alo.y; a.u[2] = alo.z; a.u[3] = alo.w;
        a.u[4] = ahi.x; a.u[5] = ahi.y; a.u[6] = ahi.z; a.u[7] = ahi.w;

#pragma unroll
        for (int t = 0; t < 4; ++t) {
            const int n = n0 + t * 16 + lanelo;
            // B fragment: lane holds 16 contiguous K of W row n (B[k,n]=W[n,k])
            FragBF b;
            const uint4* bp = (const uint4*)(Wih + (size_t)n * K + kb + half * 16);
            uint4 blo = bp[0], bhi = bp[1];
            b.u[0] = blo.x; b.u[1] = blo.y; b.u[2] = blo.z; b.u[3] = blo.w;
            b.u[4] = bhi.x; b.u[5] = bhi.y; b.u[6] = bhi.z; b.u[7] = bhi.w;

            acc[t] = __builtin_amdgcn_wmma_f32_16x16x32_bf16(
                false, a.v, false, b.v, (short)0, acc[t], false, false);
        }
    }

#pragma unroll
    for (int t = 0; t < 4; ++t) {
        const int n = n0 + t * 16 + lanelo;
#pragma unroll
        for (int r = 0; r < 8; ++r) {
            const int m = m0 + half * 8 + r;
            const size_t idx = (size_t)m * N + n;
            pre[idx] = acc[t][r] + internal[idx] + bias[n];
        }
    }
}

// ---------------------------------------------------------------------------
// Phase 2 (one time step): h_new = tanh(pre[:,t,:] + h_prev @ W_hh^T)
//   M = B = 64, N = H = 512, K = H = 512
//   16 blocks x 8 waves = 128 waves, wave = one 16x16 tile, K-loop of 16
//   WMMA ops. Block owns one 16-row h strip (staged bf16 in LDS) and 8 N
//   tiles. pre lives in d_out and is overwritten in place with the output.
// ---------------------------------------------------------------------------
__global__ void __launch_bounds__(256)
rnn_step(const unsigned short* __restrict__ hprev,   // [B, H] bf16 bits
         const unsigned short* __restrict__ Whh,     // [H, H] bf16 bits
         float* __restrict__ out,                    // [B, S, H] (pre -> tanh)
         unsigned short* __restrict__ hnext,         // [B, H] bf16 bits
         int t)
{
    constexpr int K = 512, N = 512, S = 512;
    __shared__ unsigned int lds_h[16 * LDS_STRIDE];

    const int tid    = threadIdx.x;
    const int wave   = tid >> 5;
    const int lane   = tid & 31;
    const int half   = lane >> 4;
    const int lanelo = lane & 15;
    const int m0     = (blockIdx.x >> 2) * 16;                  // batch strip
    const int n0     = (((blockIdx.x & 3) << 3) + wave) * 16;   // H tile

    // ---- cooperative stage: 16x512 bf16 (16 KB) -> padded LDS ----
    const unsigned int* hsrc = (const unsigned int*)hprev;      // 256 dw / row
    for (int q = tid; q < 4096; q += 256) {
        const int row = q >> 8;
        const int c   = q & 255;
        lds_h[row * LDS_STRIDE + c] = hsrc[(size_t)(m0 + row) * 256 + c];
    }
    __syncthreads();

    v8f acc = {0.f, 0.f, 0.f, 0.f, 0.f, 0.f, 0.f, 0.f};

    const unsigned int* alds = lds_h + lanelo * LDS_STRIDE;
    const unsigned short* brow = Whh + (size_t)(n0 + lanelo) * K;

    for (int kb = 0; kb < K; kb += 32) {
        const int di = (kb >> 1) + half * 8;
        FragBF a;
        uint4 alo = *(const uint4*)(alds + di);
        uint4 ahi = *(const uint4*)(alds + di + 4);
        a.u[0] = alo.x; a.u[1] = alo.y; a.u[2] = alo.z; a.u[3] = alo.w;
        a.u[4] = ahi.x; a.u[5] = ahi.y; a.u[6] = ahi.z; a.u[7] = ahi.w;

        FragBF b;
        const uint4* bp = (const uint4*)(brow + kb + half * 16);
        uint4 blo = bp[0], bhi = bp[1];
        b.u[0] = blo.x; b.u[1] = blo.y; b.u[2] = blo.z; b.u[3] = blo.w;
        b.u[4] = bhi.x; b.u[5] = bhi.y; b.u[6] = bhi.z; b.u[7] = bhi.w;

        acc = __builtin_amdgcn_wmma_f32_16x16x32_bf16(
            false, a.v, false, b.v, (short)0, acc, false, false);
    }

    const int n = n0 + lanelo;
#pragma unroll
    for (int r = 0; r < 8; ++r) {
        const int b = m0 + half * 8 + r;
        const size_t pidx = ((size_t)b * S + t) * (size_t)N + n;
        const float v = tanhf(acc[r] + out[pidx]);
        out[pidx] = v;
        hnext[(size_t)b * N + n] = f32_to_bf16(v);
    }
}

// ---------------------------------------------------------------------------
// Final state: state_out[b,h] = outputs[b, S-1, h]
// ---------------------------------------------------------------------------
__global__ void __launch_bounds__(256)
copy_final_state(const float* __restrict__ outputs, float* __restrict__ state_out) {
    constexpr int B = 64, S = 512, H = 512;
    int i = blockIdx.x * 256 + threadIdx.x;
    if (i < B * H) {
        int b = i / H, h = i % H;
        state_out[i] = outputs[((size_t)b * S + (S - 1)) * H + h];
    }
}

// ---------------------------------------------------------------------------
extern "C" void kernel_launch(void* const* d_in, const int* in_sizes, int n_in,
                              void* d_out, int out_size, void* d_ws, size_t ws_size,
                              hipStream_t stream)
{
    (void)in_sizes; (void)n_in; (void)out_size; (void)ws_size;
    constexpr int B = 64, S = 512, I = 512, H = 512;

    const float* input    = (const float*)d_in[0];   // [B,S,I]
    const float* internal = (const float*)d_in[1];   // [B,S,H]
    const float* state    = (const float*)d_in[2];   // [1,B,H]
    const float* w_ih     = (const float*)d_in[3];   // [H,I]
    const float* w_hh     = (const float*)d_in[4];   // [H,H]
    const float* bias     = (const float*)d_in[5];   // [H]

    float* outputs   = (float*)d_out;                      // [B,S,H]
    float* state_out = outputs + (size_t)B * S * H;        // [1,B,H]

    // workspace: bf16 weights + bf16 hidden-state ping-pong (~1.1 MB)
    unsigned short* wih_bf = (unsigned short*)d_ws;
    unsigned short* whh_bf = wih_bf + (size_t)H * I;
    unsigned short* hbuf0  = whh_bf + (size_t)H * H;
    unsigned short* hbuf1  = hbuf0 + (size_t)B * H;

    cvt_f32_bf16<<<(H * I + 255) / 256, 256, 0, stream>>>(w_ih, wih_bf, H * I);
    cvt_f32_bf16<<<(H * H + 255) / 256, 256, 0, stream>>>(w_hh, whh_bf, H * H);
    cvt_f32_bf16<<<(B * H + 255) / 256, 256, 0, stream>>>(state, hbuf0, B * H);

    // pre = input @ W_ih^T + internal + bias  -> written into outputs region
    rnn_pre_gemm<<<(B * S) / 16, 256, 0, stream>>>(input, wih_bf, internal, bias, outputs);

    // sequential recurrence: 512 small WMMA GEMM steps (graph-capturable)
    unsigned short* bufs[2] = {hbuf0, hbuf1};
    for (int t = 0; t < S; ++t) {
        rnn_step<<<16, 256, 0, stream>>>(bufs[t & 1], whh_bf, outputs,
                                         bufs[(t + 1) & 1], t);
    }

    copy_final_state<<<(B * H + 255) / 256, 256, 0, stream>>>(outputs, state_out);
}
